// DomainSpecificCapacity_77438260347449
// MI455X (gfx1250) — matrix operand order repaired and checked by reading, not compiled, not run on hardware
//
#include <hip/hip_runtime.h>

// ---------------------------------------------------------------------------
// DomainSpecificCapacity on MI455X (gfx1250, wave32, WMMA 16x16x32 f16)
//
// out[t] = emb[tok[t]] + 0.1 * W_proj[dom] @ (W_exp[dom] @ emb[tok[t]])
// (correction only for routed tokens, dom >= 0)
//
// Bandwidth-bound: base term kept exact f32; rank-32 correction computed with
// v_wmma_f32_16x16x32_f16 (f32 accumulate). One wave per 16-token tile;
// per-tile ballot skips domains not present in the tile (~2.7 of 8 active).
// ---------------------------------------------------------------------------

typedef __attribute__((ext_vector_type(16))) _Float16 v16h;
typedef __attribute__((ext_vector_type(8)))  _Float16 v8h;
typedef __attribute__((ext_vector_type(8)))  float    v8f;

#define DIMD   1024
#define EXPD   32
#define NDOM   8
#define TPT    16      // tokens per tile (one wave per tile)
#define WAVES  4       // waves per block
#define NTILES_N (DIMD / 16)

__device__ __forceinline__ v8f wmma_f16(v16h a, v16h b, v8f c) {
  return __builtin_amdgcn_wmma_f32_16x16x32_f16(
      /*neg_a=*/false, a, /*neg_b=*/false, b,
      /*c_mod=*/(short)0, c, /*reuse_a=*/false, /*reuse_b=*/false);
}

// Load 16 consecutive f32 (64B, 16B-aligned) and convert to 16 f16 (ascending K).
__device__ __forceinline__ v16h load16f_cvt(const float* __restrict__ p) {
  const float4 a = *reinterpret_cast<const float4*>(p + 0);
  const float4 b = *reinterpret_cast<const float4*>(p + 4);
  const float4 c = *reinterpret_cast<const float4*>(p + 8);
  const float4 d = *reinterpret_cast<const float4*>(p + 12);
  v16h r;
  r[0]=(_Float16)a.x;  r[1]=(_Float16)a.y;  r[2]=(_Float16)a.z;  r[3]=(_Float16)a.w;
  r[4]=(_Float16)b.x;  r[5]=(_Float16)b.y;  r[6]=(_Float16)b.z;  r[7]=(_Float16)b.w;
  r[8]=(_Float16)c.x;  r[9]=(_Float16)c.y;  r[10]=(_Float16)c.z; r[11]=(_Float16)c.w;
  r[12]=(_Float16)d.x; r[13]=(_Float16)d.y; r[14]=(_Float16)d.z; r[15]=(_Float16)d.w;
  return r;
}

// Load 8 consecutive f32 into halves [base..base+7] of a v16h.
__device__ __forceinline__ void load8f_cvt(const float* __restrict__ p, v16h& r, int base) {
  const float4 a = *reinterpret_cast<const float4*>(p + 0);
  const float4 b = *reinterpret_cast<const float4*>(p + 4);
  r[base+0]=(_Float16)a.x; r[base+1]=(_Float16)a.y; r[base+2]=(_Float16)a.z; r[base+3]=(_Float16)a.w;
  r[base+4]=(_Float16)b.x; r[base+5]=(_Float16)b.y; r[base+6]=(_Float16)b.z; r[base+7]=(_Float16)b.w;
}

__global__ __launch_bounds__(WAVES * 32)
void DomainSpecificCapacity_77438260347449_kernel(
    const int*   __restrict__ token_ids,   // [n_tok]
    const int*   __restrict__ domain_map,  // [VOCAB]
    const float* __restrict__ emb,         // [VOCAB, DIM]
    const float* __restrict__ Wexp,        // [NDOM, EXP, DIM]
    const float* __restrict__ Wproj,       // [NDOM, DIM, EXP]
    float*       __restrict__ out,         // [n_tok, DIM]
    int n_tiles)
{
  // per-wave scratch: masked, 0.1-scaled "expanded" (16 x 32) in f16, per domain
  __shared__ __align__(16) _Float16 expLds[WAVES][NDOM][TPT][EXPD]; // 32 KB
  __shared__ int tokLds[WAVES][TPT];
  __shared__ int domLds[WAVES][TPT];

  const int wave = threadIdx.x >> 5;
  const int lane = threadIdx.x & 31;
  const int m    = lane & 15;        // row / col index within a 16-wide tile
  const int half = lane >> 4;        // K-half selector per WMMA layouts

  const int tile = blockIdx.x * WAVES + wave;
  if (tile >= n_tiles) return;       // wave-uniform (tile depends only on wave)
  const int tokBase = tile * TPT;

  // Token + domain for row m (lanes L and L+16 duplicate row m -> consistent)
  const int tok   = token_ids[tokBase + m];
  const int dmine = domain_map[tok];
  if (half == 0) { tokLds[wave][m] = tok; domLds[wave][m] = dmine; }

  // Wave-uniform bitmask of domains present in this tile
  unsigned pres = 0;
  #pragma unroll
  for (int d = 0; d < NDOM; ++d)
    if (__builtin_amdgcn_ballot_w32(dmine == d)) pres |= 1u << d;

  const float* rowp = emb + (size_t)tok * DIMD;

  // ---------------- Stage 1: expanded_d = 0.1 * mask_d * (base @ Wexp[d]^T) --
  #pragma unroll 1
  for (int d = 0; d < NDOM; ++d) {
    if (!(pres & (1u << d))) continue;           // uniform branch, EXEC stays full

    v8f acc0 = {};                                // e = 0..15
    v8f acc1 = {};                                // e = 16..31
    const float* w0 = Wexp + ((size_t)d * EXPD + m)      * DIMD;  // B col n=m, tile j=0
    const float* w1 = Wexp + ((size_t)d * EXPD + 16 + m) * DIMD;  // tile j=1

    for (int kc = 0; kc < DIMD; kc += 32) {
      // A (16x32 f16): lane holds row m; K = kc + half*16 .. +15 (contiguous)
      v16h a;
      load8f_cvt(rowp + kc + half * 16,     a, 0);
      load8f_cvt(rowp + kc + half * 16 + 8, a, 8);
      // B (32x16 f16): lane holds col n=m, K = kc + half*16 .. +15 (contiguous)
      v16h b0 = load16f_cvt(w0 + kc + half * 16);
      v16h b1 = load16f_cvt(w1 + kc + half * 16);
      acc0 = wmma_f16(a, b0, acc0);
      acc1 = wmma_f16(a, b1, acc1);
    }

    // C layout: VGPR i -> row (i + 8*half), col = m (+16 for acc1).
    // Fold the 0.1 scale and the routing mask here.
    #pragma unroll
    for (int i = 0; i < 8; ++i) {
      const int row = i + 8 * half;
      const float s = (domLds[wave][row] == d) ? 0.1f : 0.0f;
      expLds[wave][d][row][m]      = (_Float16)(acc0[i] * s);
      expLds[wave][d][row][16 + m] = (_Float16)(acc1[i] * s);
    }
  }

  // ------------- Stage 2: out = base + sum_d expanded_d @ Wproj[d]^T ---------
  // Per-domain A fragments (16x32 f16) rebuilt once from LDS.
  v16h afr[NDOM] = {};
  #pragma unroll
  for (int d = 0; d < NDOM; ++d) {
    if (!(pres & (1u << d))) continue;
    const v8h lo = *reinterpret_cast<const v8h*>(&expLds[wave][d][m][half * 16]);
    const v8h hi = *reinterpret_cast<const v8h*>(&expLds[wave][d][m][half * 16 + 8]);
    v16h a;
    #pragma unroll
    for (int i = 0; i < 8; ++i) { a[i] = lo[i]; a[8 + i] = hi[i]; }
    afr[d] = a;
  }

  #pragma unroll 1
  for (int nt = 0; nt < NTILES_N; ++nt) {
    const int ncol = nt * 16 + m;
    v8f acc = {};
    #pragma unroll
    for (int d = 0; d < NDOM; ++d) {
      if (!(pres & (1u << d))) continue;
      // B (32x16): lane holds col n=ncol; K(=e) = half*16 .. +15; Wproj row-major [DIM][EXP]
      const float* wp = Wproj + (((size_t)d * DIMD + ncol) * EXPD) + half * 16;
      v16h b = load16f_cvt(wp);
      acc = wmma_f16(afr[d], b, acc);
    }
    // D layout: VGPR i -> row (i + 8*half), col ncol. base re-read hits L2.
    #pragma unroll
    for (int i = 0; i < 8; ++i) {
      const int row = i + 8 * half;
      const size_t col = (size_t)nt * 16 + m;
      const float base = emb[(size_t)tokLds[wave][row] * DIMD + col];
      out[(size_t)(tokBase + row) * DIMD + col] = base + acc[i];
    }
  }
}

extern "C" void kernel_launch(void* const* d_in, const int* in_sizes, int n_in,
                              void* d_out, int out_size, void* d_ws, size_t ws_size,
                              hipStream_t stream) {
  const int*   token_ids  = (const int*)  d_in[0];
  const int*   domain_map = (const int*)  d_in[1];
  const float* emb        = (const float*)d_in[2];
  const float* Wexp       = (const float*)d_in[3];
  const float* Wproj      = (const float*)d_in[4];
  float*       out        = (float*)d_out;

  const int n_tok   = in_sizes[0];          // B*S = 16384
  const int n_tiles = n_tok / TPT;          // 1024 tiles of 16 tokens
  const int blocks  = (n_tiles + WAVES - 1) / WAVES;

  DomainSpecificCapacity_77438260347449_kernel<<<dim3(blocks), dim3(WAVES * 32), 0, stream>>>(
      token_ids, domain_map, emb, Wexp, Wproj, out, n_tiles);
}
